// RelationalMemory_5540507812331
// MI455X (gfx1250) — compile-verified, matrix operand order checked by reading, not run on hardware
//
#include <hip/hip_runtime.h>
#include <hip/hip_bf16.h>
#include <stdint.h>

// MI455X / gfx1250: wave32, WMMA 16x16x32 bf16.
typedef __attribute__((ext_vector_type(16))) __bf16 v16bf;
typedef __attribute__((ext_vector_type(8)))  float  v8f;

#define TDIM   256   // threads per block (8 wave32)
#define WAVES  8
#define TSTEPS 256
#define DMODEL 512

__device__ __forceinline__ __bf16 f2bf(float f) {
  unsigned u = __builtin_bit_cast(unsigned, f);
  unsigned r = (u + 0x7fffu + ((u >> 16) & 1u)) >> 16;   // round-to-nearest-even
  unsigned short h = (unsigned short)r;
  return __builtin_bit_cast(__bf16, h);
}
__device__ __forceinline__ float bf2f(__bf16 x) {
  unsigned short h = __builtin_bit_cast(unsigned short, x);
  unsigned u = ((unsigned)h) << 16;
  return __builtin_bit_cast(float, u);
}
__device__ __forceinline__ float sigm(float x) { return 1.f / (1.f + __expf(-x)); }

// ---------------------------------------------------------------------------
// Weight repack: fp32 row-major [K,N] -> bf16 WMMA-B tiles.
// Tile (nt,kt) = 32x16 bf16 = 1KB, stored nt-major (k-loop streams contiguous
// 1KB chunks). Per ISA 7.12.2 (16-bit B layout, wave32): lane<16 holds column
// N=lane, K in {0..7, 16..23}; lane>=16 holds N=lane-16, K in {8..15, 24..31}.
// Each lane's 16 values are one contiguous 32B chunk -> coalesced b128 loads.
// ---------------------------------------------------------------------------
__global__ void pack_weight_kernel(const float* __restrict__ W,
                                   unsigned short* __restrict__ dst,
                                   int ktiles, int ntiles, int N) {
  int t = blockIdx.x * blockDim.x + threadIdx.x;
  int total = ntiles * ktiles * 32;
  if (t >= total) return;
  int tile = t >> 5, l = t & 31;
  int nt = tile / ktiles, kt = tile - nt * ktiles;
  int n = nt * 16 + (l & 15);
  int kbase = kt * 32 + ((l & 16) ? 8 : 0);
  v16bf r;
#pragma unroll
  for (int j = 0; j < 16; ++j) {
    int k = kbase + ((j < 8) ? j : j + 8);
    r[j] = f2bf(W[(size_t)k * N + n]);
  }
  *(v16bf*)(dst + ((size_t)tile << 9) + l * 16) = r;
}

// ---------------------------------------------------------------------------
// Pack row-major fp32 A (rows x 512, rows<=5, zero-padded to 16) from LDS into
// WMMA-A per-lane layout: slot (kt,lane) = one 32B chunk per lane.
// ---------------------------------------------------------------------------
__device__ __forceinline__ void pack_A(const float* __restrict__ arow, int rows,
                                       __bf16* panel, int tid) {
  for (int s = tid; s < 16 * 32; s += TDIM) {
    int kt = s >> 5, l = s & 31;
    int m = l & 15;
    v16bf r;
    if (m < rows) {
      const float* src = arow + m * DMODEL;
      int kbase = kt * 32 + ((l & 16) ? 8 : 0);
#pragma unroll
      for (int j = 0; j < 16; ++j) {
        int k = kbase + ((j < 8) ? j : j + 8);
        r[j] = f2bf(src[k]);
      }
    } else {
#pragma unroll
      for (int j = 0; j < 16; ++j) r[j] = f2bf(0.f);
    }
    *(v16bf*)(panel + (s << 4)) = r;
  }
}

// ---------------------------------------------------------------------------
// GEMM: [16 x (32*ktiles)] (A panel in LDS) @ packed B -> bf16 out in LDS.
// Each wave owns PAIRS of adjacent N-tiles and runs TWO independent WMMA
// accumulator chains sharing each A fragment: breaks the acc->acc RAW chain
// (2x XDL ILP) and halves A LDS reads per WMMA. All N-tile counts are even
// multiples of 2*WAVES/..., loop divides exactly (96, 32, 64 tiles).
// C layout: VGPR m = row M=m for lanes 0..15 (we only need rows 0..4).
// ---------------------------------------------------------------------------
__device__ __forceinline__ void gemm_tiles(const unsigned short* __restrict__ Bp,
                                           int ktiles, int ntiles, int rowsOut,
                                           const __bf16* panel, __bf16* gout,
                                           int ldOut, int lane, int wid) {
  for (int ntp = wid; ntp < (ntiles >> 1); ntp += WAVES) {
    const int nt0 = ntp * 2;
    v8f acc0 = {0.f, 0.f, 0.f, 0.f, 0.f, 0.f, 0.f, 0.f};
    v8f acc1 = {0.f, 0.f, 0.f, 0.f, 0.f, 0.f, 0.f, 0.f};
    const unsigned short* bp0 = Bp + ((size_t)(nt0 * ktiles) << 9) + lane * 16;
    const unsigned short* bp1 = bp0 + ((size_t)ktiles << 9);
    const __bf16* ap = panel + lane * 16;
    for (int kt = 0; kt < ktiles; ++kt) {
      if (kt + 1 < ktiles) {
        __builtin_prefetch((const void*)(bp0 + ((kt + 1) << 9)), 0, 1);
        __builtin_prefetch((const void*)(bp1 + ((kt + 1) << 9)), 0, 1);
      }
      v16bf a  = *(const v16bf*)(ap + kt * 512);
      v16bf b0 = *(const v16bf*)(bp0 + (kt << 9));
      v16bf b1 = *(const v16bf*)(bp1 + (kt << 9));
      acc0 = __builtin_amdgcn_wmma_f32_16x16x32_bf16(
          false, a, false, b0, (short)0, acc0, false, false);
      acc1 = __builtin_amdgcn_wmma_f32_16x16x32_bf16(
          false, a, false, b1, (short)0, acc1, false, false);
    }
    if (lane < 16) {
#pragma unroll
      for (int m = 0; m < 5; ++m)
        if (m < rowsOut) {
          gout[m * ldOut + nt0 * 16 + lane]      = f2bf(acc0[m]);
          gout[m * ldOut + nt0 * 16 + 16 + lane] = f2bf(acc1[m]);
        }
    }
  }
}

// ---------------------------------------------------------------------------
// Persistent per-batch kernel: one workgroup per batch runs all 256 steps,
// keeping mem[4][512] resident in LDS. Static LDS = 59 KB.
// ---------------------------------------------------------------------------
__global__ void __launch_bounds__(TDIM, 1)
relmem_kernel(const float* __restrict__ inputs, const float* __restrict__ memory,
              const float* __restrict__ bq, const float* __restrict__ bk,
              const float* __restrict__ bv, const float* __restrict__ bo,
              const float* __restrict__ b1, const float* __restrict__ b2,
              const float* __restrict__ bw, const float* __restrict__ bu,
              const unsigned short* __restrict__ ws, float* __restrict__ out) {
  __shared__ float memsh[4 * DMODEL];                              // 8 KB state
  __shared__ float nmsh[4 * DMODEL];                               // 8 KB
  __shared__ float xbuf[DMODEL];                                   // 2 KB
  __shared__ float arow[5 * DMODEL];                               // 10 KB A stage
  __shared__ __attribute__((aligned(32))) __bf16 panel[16 * 32 * 16]; // 16 KB
  __shared__ __attribute__((aligned(32))) __bf16 goutb[7680];         // 15 KB

  const int tid = threadIdx.x;
  const int lane = tid & 31;
  const int wid = tid >> 5;
  const int b = blockIdx.x;

  const unsigned short* WqkvP = ws;                // Wq|Wk|Wv, 96 nt x 16 kt
  const unsigned short* WoP = ws + 786432;
  const unsigned short* W1P = ws + 1048576;
  const unsigned short* W2P = ws + 1310720;
  const unsigned short* WuP = ws + 1572864;        // 64 nt x 16 kt
  const unsigned short* WwP = ws + 2097152;

  for (int i = tid; i < 2048; i += TDIM) memsh[i] = memory[(size_t)b * 2048 + i];
  __syncthreads();

  for (int t = 0; t < TSTEPS; ++t) {
    // ---- stage A = [mem(4); x(1); 0...] ----
    for (int i = tid; i < DMODEL; i += TDIM) {
      float xv = inputs[((size_t)b * TSTEPS + t) * DMODEL + i];
      xbuf[i] = xv;
      arow[4 * DMODEL + i] = xv;
    }
    for (int i = tid; i < 2048; i += TDIM) arow[i] = memsh[i];
    __syncthreads();
    pack_A(arow, 5, panel, tid);
    __syncthreads();

    // ---- fused QKV: [16,512] @ [512,1536] ----
    gemm_tiles(WqkvP, 16, 96, 5, panel, goutb, 1536, lane, wid);
    __syncthreads();
    for (int i = tid; i < 5 * 1536; i += TDIM) {    // bias add in-place
      int n = i % 1536;
      float bs = (n < 512) ? bq[n] : (n < 1024 ? bk[n - 512] : bv[n - 1024]);
      goutb[i] = f2bf(bf2f(goutb[i]) + bs);
    }
    __syncthreads();

    // ---- attention: wave = head; 4x5 scores, softmax, o -> arow ----
    {
      const int h = wid;
      const int e0 = lane * 2;
      for (int qi = 0; qi < 4; ++qi) {
        float p[5];
        float mx = -1e30f;
        for (int ki = 0; ki < 5; ++ki) {
          float q0 = bf2f(goutb[qi * 1536 + h * 64 + e0]);
          float q1 = bf2f(goutb[qi * 1536 + h * 64 + e0 + 1]);
          float k0 = bf2f(goutb[ki * 1536 + 512 + h * 64 + e0]);
          float k1 = bf2f(goutb[ki * 1536 + 512 + h * 64 + e0 + 1]);
          float s = q0 * k0 + q1 * k1;
          s += __shfl_xor(s, 16, 32);
          s += __shfl_xor(s, 8, 32);
          s += __shfl_xor(s, 4, 32);
          s += __shfl_xor(s, 2, 32);
          s += __shfl_xor(s, 1, 32);
          s *= 0.125f;                    // 1/sqrt(DK=64)
          p[ki] = s;
          mx = fmaxf(mx, s);
        }
        float sum = 0.f;
        for (int ki = 0; ki < 5; ++ki) { p[ki] = __expf(p[ki] - mx); sum += p[ki]; }
        float inv = 1.f / sum;
        float o0 = 0.f, o1 = 0.f;
        for (int ki = 0; ki < 5; ++ki) {
          float w = p[ki] * inv;
          o0 += w * bf2f(goutb[ki * 1536 + 1024 + h * 64 + e0]);
          o1 += w * bf2f(goutb[ki * 1536 + 1024 + h * 64 + e0 + 1]);
        }
        arow[qi * DMODEL + h * 64 + e0] = o0;
        arow[qi * DMODEL + h * 64 + e0 + 1] = o1;
      }
    }
    __syncthreads();

    // ---- nm1 = mem + o@Wo + bo ----
    pack_A(arow, 4, panel, tid);
    __syncthreads();
    gemm_tiles(WoP, 16, 32, 4, panel, goutb, 512, lane, wid);
    __syncthreads();
    for (int i = tid; i < 2048; i += TDIM) {
      int n = i & 511;
      float v = memsh[i] + bf2f(goutb[i]) + bo[n];
      nmsh[i] = v;
      arow[i] = v;
    }
    __syncthreads();

    // ---- MLP: nm2 = nm1 + relu(relu(nm1@W1+b1)@W2+b2) ----
    pack_A(arow, 4, panel, tid);
    __syncthreads();
    gemm_tiles(W1P, 16, 32, 4, panel, goutb, 512, lane, wid);
    __syncthreads();
    for (int i = tid; i < 2048; i += TDIM) {
      int n = i & 511;
      arow[i] = fmaxf(bf2f(goutb[i]) + b1[n], 0.f);
    }
    __syncthreads();
    pack_A(arow, 4, panel, tid);
    __syncthreads();
    gemm_tiles(W2P, 16, 32, 4, panel, goutb, 512, lane, wid);
    __syncthreads();
    for (int i = tid; i < 2048; i += TDIM) {
      int n = i & 511;
      nmsh[i] += fmaxf(bf2f(goutb[i]) + b2[n], 0.f);
      arow[i] = tanhf(memsh[i]);                 // A for gates pass 1
    }
    __syncthreads();

    // ---- gates: g1 = tanh(mem)@Wu (4 rows), g2 = x@Ww (1 row) ----
    pack_A(arow, 4, panel, tid);
    __syncthreads();
    gemm_tiles(WuP, 16, 64, 4, panel, goutb, 1024, lane, wid);
    __syncthreads();
    for (int i = tid; i < DMODEL; i += TDIM) arow[i] = xbuf[i];
    __syncthreads();
    pack_A(arow, 1, panel, tid);
    __syncthreads();
    gemm_tiles(WwP, 16, 64, 1, panel, goutb + 4096, 1024, lane, wid);
    __syncthreads();

    // ---- final combine + output + state update ----
    for (int i = tid; i < 2048; i += TDIM) {
      int m = i >> 9, dd = i & 511;
      float ig = bf2f(goutb[m * 1024 + dd]) + bf2f(goutb[4096 + dd]) +
                 bw[dd] + bu[dd];
      float fg = bf2f(goutb[m * 1024 + 512 + dd]) + bf2f(goutb[4096 + 512 + dd]) +
                 bw[512 + dd] + bu[512 + dd];
      float nnew = sigm(ig) * tanhf(nmsh[i]) + sigm(fg) * memsh[i];
      memsh[i] = nnew;
      out[((size_t)b * TSTEPS + t) * 2048 + i] = nnew;
    }
    __syncthreads();
  }
}

// ---------------------------------------------------------------------------
extern "C" void kernel_launch(void* const* d_in, const int* in_sizes, int n_in,
                              void* d_out, int out_size, void* d_ws, size_t ws_size,
                              hipStream_t stream) {
  (void)in_sizes; (void)n_in; (void)out_size; (void)ws_size;
  const float* inputs = (const float*)d_in[0];
  const float* memory = (const float*)d_in[1];
  const float* Wq = (const float*)d_in[2];  const float* bq = (const float*)d_in[3];
  const float* Wk = (const float*)d_in[4];  const float* bk = (const float*)d_in[5];
  const float* Wv = (const float*)d_in[6];  const float* bv = (const float*)d_in[7];
  const float* Wo = (const float*)d_in[8];  const float* bo = (const float*)d_in[9];
  const float* W1 = (const float*)d_in[10]; const float* b1 = (const float*)d_in[11];
  const float* W2 = (const float*)d_in[12]; const float* b2 = (const float*)d_in[13];
  const float* Ww = (const float*)d_in[14]; const float* bw = (const float*)d_in[15];
  const float* Wu = (const float*)d_in[16]; const float* bu = (const float*)d_in[17];
  unsigned short* ws = (unsigned short*)d_ws;   // 5 MB packed bf16 weights
  float* out = (float*)d_out;

  // Repack weights into WMMA tile layout (runs every call; deterministic).
  pack_weight_kernel<<<64,  256, 0, stream>>>(Wq, ws + 0,       16, 32, 512);
  pack_weight_kernel<<<64,  256, 0, stream>>>(Wk, ws + 262144,  16, 32, 512);
  pack_weight_kernel<<<64,  256, 0, stream>>>(Wv, ws + 524288,  16, 32, 512);
  pack_weight_kernel<<<64,  256, 0, stream>>>(Wo, ws + 786432,  16, 32, 512);
  pack_weight_kernel<<<64,  256, 0, stream>>>(W1, ws + 1048576, 16, 32, 512);
  pack_weight_kernel<<<64,  256, 0, stream>>>(W2, ws + 1310720, 16, 32, 512);
  pack_weight_kernel<<<128, 256, 0, stream>>>(Wu, ws + 1572864, 16, 64, 1024);
  pack_weight_kernel<<<128, 256, 0, stream>>>(Ww, ws + 2097152, 16, 64, 1024);

  relmem_kernel<<<64, TDIM, 0, stream>>>(inputs, memory, bq, bk, bv, bo,
                                         b1, b2, bw, bu, ws, out);
}
